// MambaAudioVisualFusion_87282325389777
// MI455X (gfx1250) — compile-verified, matrix operand order checked by reading, not run on hardware
//
#include <hip/hip_runtime.h>
#include <math.h>

// ---------------------------------------------------------------------------
// Types for CDNA5 WMMA
// ---------------------------------------------------------------------------
typedef __attribute__((ext_vector_type(16))) _Float16 v16h;
typedef __attribute__((ext_vector_type(8)))  _Float16 v8h;
typedef __attribute__((ext_vector_type(8)))  float    v8f;

#define LSEQ   1024
#define NBATCH 4
#define MTOK   4096      // NBATCH * LSEQ tokens
#define DSTATE 16

// ---------------------------------------------------------------------------
// Workspace layout (in floats)
// ---------------------------------------------------------------------------
static const size_t WS_S     = 8388608ull;                 // 4096*2048 scratch unit
static const size_t OFF0     = 0;
static const size_t OFF1     = 1*WS_S;
static const size_t OFF2     = 2*WS_S;
static const size_t OFF3     = 3*WS_S;
static const size_t OFF4     = 4*WS_S;
static const size_t OFF_XDBL = 5*WS_S;                     // 4096*32
static const size_t OFF_A    = OFF_XDBL + 131072;          // 4096*512 audio residual
static const size_t OFF_V    = OFF_A    + 2097152;         // 4096*512 video residual
static const size_t OFF_FUSED= OFF_V    + 2097152;         // 4096*1024 fused residual
static const size_t OFF_ALIGN= OFF_FUSED+ 4194304;         // 4*1024*256 head-mean probs
static const size_t WS_TOTAL = OFF_ALIGN+ 1048576;         // floats

__device__ __forceinline__ float sigmoidf_(float x) { return 1.f / (1.f + __expf(-x)); }

// ---------------------------------------------------------------------------
// Generic WMMA GEMM: C = act(scale * A(MxK) @ W(KxN) + bias) [+= C]
//   WT=true: W operand given row-major as (N x K) (i.e. B = W^T)
//   ACT: 0 = none, 1 = softplus
// Block: 512 threads = 16 waves; block tile 128x128; each wave a 32x32 tile
// (2 A-fragments x 2 B-fragments -> 4 v_wmma per K-step, fragments reused).
// ---------------------------------------------------------------------------
template<bool WT, bool ACCUM, int ACT>
__global__ void __launch_bounds__(512)
gemm_wmma_kernel(const float* __restrict__ A, int lda,
                 const float* __restrict__ W, int ldw,
                 const float* __restrict__ bias,
                 float* __restrict__ C, int ldc,
                 int M, int N, int K, float scale)
{
    __shared__ __align__(16) _Float16 As[128][40];   // [m][k], padded rows
    __shared__ __align__(16) _Float16 Bs[128][40];   // [n][k], padded rows

    const int t    = threadIdx.x;
    const int wave = t >> 5;
    const int lane = t & 31;
    const int wm   = (wave & 3) << 5;   // wave m-offset in block tile (0..96)
    const int wn   = (wave >> 2) << 5;  // wave n-offset in block tile (0..96)
    const int m0   = blockIdx.y * 128;
    const int n0   = blockIdx.x * 128;

    const int lsel = lane >> 4;         // 0/1 half-wave
    const int lrow = lane & 15;
    const int kb0  = lsel * 8;          // K {0..7} or {8..15}
    const int kb1  = 16 + lsel * 8;     // K {16..23} or {24..31}

    v8f acc[2][2];
    #pragma unroll
    for (int i = 0; i < 2; ++i)
        #pragma unroll
        for (int j = 0; j < 2; ++j)
            acc[i][j] = (v8f){0.f,0.f,0.f,0.f,0.f,0.f,0.f,0.f};

    for (int k0 = 0; k0 < K; k0 += 32) {
        // prefetch next K-tile while this one is computed (global_prefetch_b8)
        if (k0 + 32 < K) {
            const int pm = m0 + (t >> 3);
            if (pm < M) __builtin_prefetch(A + (size_t)pm * lda + k0 + 32 + ((t & 7) << 2), 0, 1);
            if (WT) {
                const int pn = n0 + (t >> 3);
                if (pn < N) __builtin_prefetch(W + (size_t)pn * ldw + k0 + 32 + ((t & 7) << 2), 0, 1);
            } else {
                const int pk = k0 + 32 + (t >> 4);
                __builtin_prefetch(W + (size_t)pk * ldw + n0 + ((t & 15) << 2), 0, 1);
            }
        }

        // A tile 128x32 -> f16 LDS (each thread: two float4 rows)
        #pragma unroll
        for (int it = 0; it < 2; ++it) {
            const int r = (t >> 3) + 64 * it;       // 0..127
            const int c = (t & 7) << 2;             // 0..28
            const int gm = m0 + r;
            float4 av = make_float4(0.f, 0.f, 0.f, 0.f);
            if (gm < M) av = *(const float4*)(A + (size_t)gm * lda + k0 + c);
            As[r][c+0] = (_Float16)av.x;  As[r][c+1] = (_Float16)av.y;
            As[r][c+2] = (_Float16)av.z;  As[r][c+3] = (_Float16)av.w;
        }
        if (WT) { // W given as (N x K) row-major: copy rows directly into Bs[n][k]
            #pragma unroll
            for (int it = 0; it < 2; ++it) {
                const int bn = (t >> 3) + 64 * it;  // 0..127
                const int bk = (t & 7) << 2;
                const int gn = n0 + bn;
                float4 wv = make_float4(0.f, 0.f, 0.f, 0.f);
                if (gn < N) wv = *(const float4*)(W + (size_t)gn * ldw + k0 + bk);
                Bs[bn][bk+0] = (_Float16)wv.x;  Bs[bn][bk+1] = (_Float16)wv.y;
                Bs[bn][bk+2] = (_Float16)wv.z;  Bs[bn][bk+3] = (_Float16)wv.w;
            }
        } else {  // W given as (K x N): transpose into Bs[n][k] (scalar, guarded)
            #pragma unroll
            for (int it = 0; it < 2; ++it) {
                const int wrk = t >> 4;                        // 0..31
                const int wcn = ((t & 15) << 2) + 64 * it;     // 0..124
                const int gk  = k0 + wrk;
                #pragma unroll
                for (int j = 0; j < 4; ++j) {
                    const int gn = n0 + wcn + j;
                    float v = 0.f;
                    if (gn < N) v = W[(size_t)gk * ldw + gn];
                    Bs[wcn+j][wrk] = (_Float16)v;
                }
            }
        }
        __syncthreads();

        // fragment loads (two 16B ds reads each) per documented 16-bit layouts
        v16h af[2], bf[2];
        #pragma unroll
        for (int i = 0; i < 2; ++i) {
            const int arow = wm + i*16 + lrow;
            v8h a0 = *(const v8h*)&As[arow][kb0];
            v8h a1 = *(const v8h*)&As[arow][kb1];
            af[i] = __builtin_shufflevector(a0, a1, 0,1,2,3,4,5,6,7,8,9,10,11,12,13,14,15);
            const int brow = wn + i*16 + lrow;
            v8h b0 = *(const v8h*)&Bs[brow][kb0];
            v8h b1 = *(const v8h*)&Bs[brow][kb1];
            bf[i] = __builtin_shufflevector(b0, b1, 0,1,2,3,4,5,6,7,8,9,10,11,12,13,14,15);
        }
        #pragma unroll
        for (int i = 0; i < 2; ++i)
            #pragma unroll
            for (int j = 0; j < 2; ++j)
                acc[i][j] = __builtin_amdgcn_wmma_f32_16x16x32_f16(
                                false, af[i], false, bf[j], (short)0, acc[i][j],
                                false, false);
        __syncthreads();
    }

    // Epilogue: VGPR r, lanes 0-15 -> row r, lanes 16-31 -> row 8+r; col = lrow
    #pragma unroll
    for (int fj = 0; fj < 2; ++fj) {
        const int col = n0 + wn + fj*16 + lrow;
        if (col < N) {
            const float bvv = bias ? bias[col] : 0.f;
            #pragma unroll
            for (int fi = 0; fi < 2; ++fi) {
                #pragma unroll
                for (int r = 0; r < 8; ++r) {
                    const int row = m0 + wm + fi*16 + lsel*8 + r;
                    if (row < M) {
                        float v = acc[fi][fj][r] * scale + bvv;
                        if (ACT == 1) v = (v > 20.f) ? v : log1pf(__expf(v));
                        float* cp = C + (size_t)row * ldc + col;
                        if (ACCUM) v += *cp;
                        *cp = v;
                    }
                }
            }
        }
    }
}

// ---------------------------------------------------------------------------
// Elementwise / reduction kernels
// ---------------------------------------------------------------------------
__global__ void __launch_bounds__(256)
zero_kernel(float* __restrict__ p, size_t n)
{
    size_t i = (size_t)blockIdx.x * 256 + threadIdx.x;
    if (i < n) p[i] = 0.f;
}

__global__ void __launch_bounds__(256)
rmsnorm_kernel(const float* __restrict__ X, const float* __restrict__ w,
               float* __restrict__ Y, int d)
{
    __shared__ float red[256];
    const size_t row = blockIdx.x;
    const float* x = X + row * d;
    float ss = 0.f;
    for (int i = threadIdx.x; i < d; i += 256) { float v = x[i]; ss += v * v; }
    red[threadIdx.x] = ss; __syncthreads();
    for (int s = 128; s > 0; s >>= 1) {
        if (threadIdx.x < s) red[threadIdx.x] += red[threadIdx.x + s];
        __syncthreads();
    }
    const float sc = rsqrtf(red[0] / (float)d + 1.1920929e-7f);
    for (int i = threadIdx.x; i < d; i += 256) Y[row * d + i] = x[i] * sc * w[i];
}

// depthwise causal conv (K=4) fused with SiLU
__global__ void __launch_bounds__(256)
conv_silu_kernel(const float* __restrict__ x, const float* __restrict__ cw,
                 const float* __restrict__ cb, float* __restrict__ y,
                 int M, int DI, int L)
{
    size_t i = (size_t)blockIdx.x * 256 + threadIdx.x;
    if (i >= (size_t)M * DI) return;
    const int dch = (int)(i % DI);
    const int m   = (int)(i / DI);
    const int l   = m % L;
    const float* wv = cw + dch * 4;
    float s = cb[dch];
    #pragma unroll
    for (int j = 0; j < 4; ++j) {
        const int ls = l - 3 + j;
        if (ls >= 0) s += wv[j] * x[(size_t)(m - 3 + j) * DI + dch];
    }
    y[i] = s * sigmoidf_(s);
}

__global__ void __launch_bounds__(256)
mulsilu_kernel(float* __restrict__ o, const float* __restrict__ a,
               const float* __restrict__ g, size_t n)
{
    size_t i = (size_t)blockIdx.x * 256 + threadIdx.x;
    if (i >= n) return;
    const float gv = g[i];
    o[i] = a[i] * gv * sigmoidf_(gv);
}

__global__ void __launch_bounds__(256)
addmulD_kernel(float* __restrict__ y, const float* __restrict__ xs,
               const float* __restrict__ D, size_t n, int DI)
{
    size_t i = (size_t)blockIdx.x * 256 + threadIdx.x;
    if (i >= n) return;
    y[i] += xs[i] * D[i % DI];
}

__global__ void __launch_bounds__(256)
concat_kernel(const float* __restrict__ a, const float* __restrict__ v,
              float* __restrict__ f, int M)
{
    size_t i = (size_t)blockIdx.x * 256 + threadIdx.x;
    if (i >= (size_t)M * 1024) return;
    const int m = (int)(i / 1024);
    const int c = (int)(i % 1024);
    f[i] = (c < 512) ? a[(size_t)m * 512 + c] : v[(size_t)m * 512 + (c - 512)];
}

// selective scan: one thread per (b, channel); state N=16 in registers
__global__ void __launch_bounds__(256)
scan_kernel(const float* __restrict__ u, const float* __restrict__ delta,
            const float* __restrict__ xdbl, const float* __restrict__ A_log,
            float* __restrict__ y, int L, int DI)
{
    const int t = blockIdx.x * 256 + threadIdx.x;
    const int b = t / DI;
    const int d = t % DI;
    float A[DSTATE], s[DSTATE];
    #pragma unroll
    for (int n = 0; n < DSTATE; ++n) { A[n] = -__expf(A_log[n]); s[n] = 0.f; }
    const size_t base = (size_t)b * L;
    for (int l = 0; l < L; ++l) {
        const size_t tok = base + l;
        const float dl = delta[tok * DI + d];
        const float uu = u[tok * DI + d];
        const float* bm = xdbl + tok * 32;
        float yy = 0.f;
        #pragma unroll
        for (int n = 0; n < DSTATE; ++n) {
            const float dA = __expf(dl * A[n]);
            s[n] = dA * s[n] + dl * bm[n] * uu;
            yy += s[n] * bm[16 + n];
        }
        y[tok * DI + d] = yy;
    }
}

// softmax over Lk=256 per (b,h,lq) row; accumulates head-mean into align
__global__ void __launch_bounds__(128)
softmax_align_kernel(float* __restrict__ S, float* __restrict__ align)
{
    const int wave = threadIdx.x >> 5;
    const int lane = threadIdx.x & 31;
    const int r  = blockIdx.x * 4 + wave;     // 0..32767 = (b*8+h)*1024 + lq
    const int b  = r >> 13;
    const int lq = r & 1023;
    float* row = S + (size_t)r * 256;
    float v[8];
    float mx = -3.4e38f;
    #pragma unroll
    for (int j = 0; j < 8; ++j) { v[j] = row[lane + 32*j]; mx = fmaxf(mx, v[j]); }
    for (int o = 16; o >= 1; o >>= 1) mx = fmaxf(mx, __shfl_xor(mx, o));
    float sum = 0.f;
    #pragma unroll
    for (int j = 0; j < 8; ++j) { v[j] = __expf(v[j] - mx); sum += v[j]; }
    for (int o = 16; o >= 1; o >>= 1) sum += __shfl_xor(sum, o);
    const float inv = 1.f / sum;
    float* al = align + ((size_t)(b * 1024 + lq)) * 256;
    #pragma unroll
    for (int j = 0; j < 8; ++j) {
        const float p = v[j] * inv;
        row[lane + 32*j] = p;
        atomicAdd(&al[lane + 32*j], p * 0.125f);   // mean over 8 heads
    }
}

// score[b] = sigmoid( mean_lq( max_lk align ) / temp )
__global__ void __launch_bounds__(256)
score_kernel(const float* __restrict__ align, const float* __restrict__ temp,
             float* __restrict__ out)
{
    __shared__ float red[256];
    const int b = blockIdx.x, t = threadIdx.x;
    float acc = 0.f;
    for (int lq = t; lq < 1024; lq += 256) {
        const float* row = align + ((size_t)(b * 1024 + lq)) * 256;
        float mx = row[0];
        for (int j = 1; j < 256; ++j) mx = fmaxf(mx, row[j]);
        acc += mx;
    }
    red[t] = acc; __syncthreads();
    for (int s = 128; s > 0; s >>= 1) { if (t < s) red[t] += red[t + s]; __syncthreads(); }
    if (t == 0) {
        const float m = red[0] * (1.f / 1024.f);
        out[b] = sigmoidf_(m / temp[0]);
    }
}

// ---------------------------------------------------------------------------
// Host-side helpers
// ---------------------------------------------------------------------------
static inline unsigned egrid(size_t n) { return (unsigned)((n + 255) / 256); }

static void launch_gemm(hipStream_t st, const float* A, int lda, const float* W, int ldw,
                        const float* bias, float* C, int ldc, int M, int N, int K,
                        float scale, bool softplus, bool accum, bool wt)
{
    dim3 g((unsigned)((N + 127) / 128), (unsigned)((M + 127) / 128)), b(512);
    if (wt)            gemm_wmma_kernel<true , false, 0><<<g, b, 0, st>>>(A, lda, W, ldw, bias, C, ldc, M, N, K, scale);
    else if (accum)    gemm_wmma_kernel<false, true , 0><<<g, b, 0, st>>>(A, lda, W, ldw, bias, C, ldc, M, N, K, scale);
    else if (softplus) gemm_wmma_kernel<false, false, 1><<<g, b, 0, st>>>(A, lda, W, ldw, bias, C, ldc, M, N, K, scale);
    else               gemm_wmma_kernel<false, false, 0><<<g, b, 0, st>>>(A, lda, W, ldw, bias, C, ldc, M, N, K, scale);
}

// Param leaf order inside one mamba block (jax dict flatten = sorted keys):
//  +0 in_proj.w  +1 norm_w  +2 out_proj.w  +3 ssm.A_log  +4 ssm.D  +5 ssm.conv_b
//  +6 ssm.conv_w +7 ssm.dt_proj.b +8 ssm.dt_proj.w +9 ssm.in_proj.b +10 ssm.in_proj.w
//  +11 ssm.out_proj.b +12 ssm.out_proj.w +13 ssm.x_proj.b +14 ssm.x_proj.w
static void run_mamba_block(hipStream_t st, void* const* d_in, int pb,
                            float* X, int d, float* ws)
{
    const int di = 2 * d;
    const int M  = MTOK;
    auto P = [&](int i) { return (const float*)d_in[i]; };

    float* xnorm = ws + OFF2;        // consumed before xsraw is written here
    float* xssm  = ws + OFF0;        // later reused as y
    float* gate  = ws + OFF1;
    float* xsraw = ws + OFF2;        // later reused as delta
    float* res   = ws + OFF3;
    float* xs    = ws + OFF4;        // later reused as ssm_out
    float* xdbl  = ws + OFF_XDBL;
    const size_t n = (size_t)M * di;

    rmsnorm_kernel<<<M, 256, 0, st>>>(X, P(pb + 1), xnorm, d);
    // block in_proj (d -> 4d), split into x_ssm | gate
    launch_gemm(st, xnorm, d, P(pb + 0),       4*d, nullptr, xssm, di, M, di, d, 1.f, false, false, false);
    launch_gemm(st, xnorm, d, P(pb + 0) + di,  4*d, nullptr, gate, di, M, di, d, 1.f, false, false, false);
    // ssm in_proj (di -> 2di), split into xs_raw | res
    launch_gemm(st, xssm, di, P(pb + 10),       2*di, P(pb + 9),      xsraw, di, M, di, di, 1.f, false, false, false);
    launch_gemm(st, xssm, di, P(pb + 10) + di,  2*di, P(pb + 9) + di, res,   di, M, di, di, 1.f, false, false, false);
    // causal depthwise conv + SiLU
    conv_silu_kernel<<<egrid(n), 256, 0, st>>>(xsraw, P(pb + 6), P(pb + 5), xs, M, di, LSEQ);
    // x_dbl = x_proj(xs) : (M,32) = [B | C]
    launch_gemm(st, xs, di, P(pb + 14), 2*DSTATE, P(pb + 13), xdbl, 2*DSTATE, M, 2*DSTATE, di, 1.f, false, false, false);
    // delta = softplus(dt_proj(xs))   (fused epilogue; overwrites xs_raw)
    launch_gemm(st, xs, di, P(pb + 8), di, P(pb + 7), xsraw, di, M, di, di, 1.f, true, false, false);
    // selective scan -> y (overwrites x_ssm)
    scan_kernel<<<(NBATCH * di) / 256, 256, 0, st>>>(xs, xsraw, xdbl, P(pb + 3), xssm, LSEQ, di);
    // y += xs*D ; y *= silu(res)
    addmulD_kernel<<<egrid(n), 256, 0, st>>>(xssm, xs, P(pb + 4), n, di);
    mulsilu_kernel<<<egrid(n), 256, 0, st>>>(xssm, xssm, res, n);
    // ssm out_proj -> (overwrites xs)
    launch_gemm(st, xssm, di, P(pb + 12), di, P(pb + 11), xs, di, M, di, di, 1.f, false, false, false);
    // * silu(gate), then residual: X += gated @ block.out_proj
    mulsilu_kernel<<<egrid(n), 256, 0, st>>>(xs, xs, gate, n);
    launch_gemm(st, xs, di, P(pb + 2), d, nullptr, X, d, M, d, di, 1.f, false, true, false);
}

// ---------------------------------------------------------------------------
// Top-level driver
// Param flatten order (jax tree, dicts sorted by key):
//  0 audio_features  1 video_features  2 alignment_temp
//  3..47   audio_layers[0..2]  (15 leaves each)
//  48 audio_proj.b  49 audio_proj.w
//  50..94  cross_layers[0..2]
//  95 fusion_proj.b  96 fusion_proj.w  97 ln_w
//  98 mha.k.b  99 mha.k.w  100 mha.o.b  101 mha.o.w
//  102 mha.q.b 103 mha.q.w 104 mha.v.b  105 mha.v.w
//  106..150 video_layers[0..2]
//  151 video_proj.b  152 video_proj.w
// ---------------------------------------------------------------------------
extern "C" void kernel_launch(void* const* d_in, const int* in_sizes, int n_in,
                              void* d_out, int out_size, void* d_ws, size_t ws_size,
                              hipStream_t stream)
{
    (void)in_sizes; (void)n_in; (void)out_size;
    if (ws_size < WS_TOTAL * sizeof(float)) return;
    float* ws  = (float*)d_ws;
    float* out = (float*)d_out;
    auto P = [&](int i) { return (const float*)d_in[i]; };

    float* a_buf = ws + OFF_A;      // audio residual stream  (4096 x 512)
    float* v_buf = ws + OFF_V;      // video residual stream  (4096 x 512)
    float* fused = ws + OFF_FUSED;  // fused residual stream  (4096 x 1024)
    float* align = ws + OFF_ALIGN;  // head-mean probs (4,1024,256)
    float* q    = ws + OFF0;
    float* kbuf = ws + OFF1;
    float* vp   = ws + OFF2;        // video_proj, then reused as attn pre-O out
    float* vbuf = ws + OFF3;
    float* sbuf = ws + OFF4;        // attention scores/probs (32 x 1024 x 256)

    // input projections
    launch_gemm(stream, P(0), 512, P(49),  512, P(48),  a_buf, 512, MTOK, 512, 512, 1.f, false, false, false);
    launch_gemm(stream, P(1), 256, P(152), 512, P(151), vp,    512, 1024, 512, 256, 1.f, false, false, false);

    // Q/K/V
    launch_gemm(stream, a_buf, 512, P(103), 512, P(102), q,    512, MTOK, 512, 512, 1.f, false, false, false);
    launch_gemm(stream, vp,    512, P(99),  512, P(98),  kbuf, 512, 1024, 512, 512, 1.f, false, false, false);
    launch_gemm(stream, vp,    512, P(105), 512, P(104), vbuf, 512, 1024, 512, 512, 1.f, false, false, false);

    zero_kernel<<<egrid(1048576), 256, 0, stream>>>(align, 1048576);

    // scores = (Q @ K^T) / sqrt(64), per (b,h) — WT path (K rows are (Lk x hd))
    for (int b = 0; b < NBATCH; ++b)
        for (int h = 0; h < 8; ++h)
            launch_gemm(stream,
                        q    + (size_t)b * 1024 * 512 + h * 64, 512,
                        kbuf + (size_t)b * 256  * 512 + h * 64, 512,
                        nullptr,
                        sbuf + (size_t)(b * 8 + h) * 262144, 256,
                        1024, 256, 64, 0.125f, false, false, true);

    softmax_align_kernel<<<8192, 128, 0, stream>>>(sbuf, align);

    // out = P @ V per (b,h) -> attn pre-O (reuses vp region)
    for (int b = 0; b < NBATCH; ++b)
        for (int h = 0; h < 8; ++h)
            launch_gemm(stream,
                        sbuf + (size_t)(b * 8 + h) * 262144, 256,
                        vbuf + (size_t)b * 256 * 512 + h * 64, 512,
                        nullptr,
                        vp   + (size_t)b * 1024 * 512 + h * 64, 512,
                        1024, 64, 256, 1.f, false, false, false);

    // O projection -> video_aligned residual stream
    launch_gemm(stream, vp, 512, P(101), 512, P(100), v_buf, 512, MTOK, 512, 512, 1.f, false, false, false);

    // alignment score per batch -> tail of d_out
    score_kernel<<<NBATCH, 256, 0, stream>>>(align, P(2), out + 2097152);

    // Mamba chains
    for (int i = 0; i < 3; ++i) run_mamba_block(stream, d_in, 3   + 15 * i, a_buf, 512,  ws);
    for (int i = 0; i < 3; ++i) run_mamba_block(stream, d_in, 106 + 15 * i, v_buf, 512,  ws);
    concat_kernel<<<egrid((size_t)MTOK * 1024), 256, 0, stream>>>(a_buf, v_buf, fused, MTOK);
    for (int i = 0; i < 3; ++i) run_mamba_block(stream, d_in, 50  + 15 * i, fused, 1024, ws);

    // fusion projection + final RMSNorm -> d_out
    launch_gemm(stream, fused, 1024, P(96), 512, P(95), ws + OFF0, 512, MTOK, 512, 1024, 1.f, false, false, false);
    rmsnorm_kernel<<<MTOK, 256, 0, stream>>>(ws + OFF0, P(97), out, 512);
}
